// LieAttention_28750511079446
// MI455X (gfx1250) — matrix operand order repaired and checked by reading, not compiled, not run on hardware
//
#include <hip/hip_runtime.h>
#include <math.h>

// LieAttention fused kernel for gfx1250 (MI455X, wave32, WMMA).
// B=8, L=S=1024, H=16, E=64.
// Outputs concatenated in d_out: V (B,L,H,E) | A (B,H,L,S) | entropy (B,H,L).

#define DIM_B 8
#define DIM_L 1024
#define DIM_S 1024
#define DIM_H 16
#define DIM_E 64
#define LBLK  128   // l-rows per workgroup (8 waves x 16 rows)
#define SBLK  32    // s-rows per inner iteration (2 sub-tiles of 16)

typedef __attribute__((ext_vector_type(16))) __bf16 v16bf;
typedef __attribute__((ext_vector_type(8)))  float  v8f;

union OpU { v16bf v; uint4 u[2]; };

__device__ __forceinline__ unsigned short f2bf(float f) {
  unsigned u = __float_as_uint(f);
  u += 0x7FFFu + ((u >> 16) & 1u);     // round-to-nearest-even
  return (unsigned short)(u >> 16);
}

// Hardware tanh (v_tanh_f32 on gfx1250) with branchless trans-op fallback.
__device__ __forceinline__ float fast_tanh(float x) {
#if __has_builtin(__builtin_amdgcn_tanhf)
  return __builtin_amdgcn_tanhf(x);
#elif __has_builtin(__builtin_amdgcn_tanh_f32)
  return __builtin_amdgcn_tanh_f32(x);
#else
  // tanh(x) = 1 - 2/(exp2(2*log2e*x)+1); saturates correctly at +/-inf inputs.
  const float e = __builtin_amdgcn_exp2f(x * 2.88539008177f);
  return 1.0f - 2.0f * __builtin_amdgcn_rcpf(e + 1.0f);
#endif
}

// att = (1/8) * gelu_exact(t) for t in [-1,1]:
// erf(u), |u| <= 0.7071, via 5-term odd Taylor poly (abs err ~2e-5).
__device__ __forceinline__ float att_from_t(float t) {
  const float u  = 0.70710678f * t;
  const float u2 = u * u;
  float p = fmaf(u2, 0.00522397762f, -0.0268661706f);
  p = fmaf(u2, p, 0.112837917f);
  p = fmaf(u2, p, -0.376126389f);
  p = fmaf(u2, p, 1.12837917f);
  const float erfu = u * p;
  return 0.0625f * t * (1.0f + erfu);   // 0.125 * 0.5 * t * (1 + erf)
}

// A-operand (16xK, 16-bit elems). ISA layout: lanes 0-15 hold K {kb..kb+7, kb+16..kb+23},
// lanes 16-31 hold K {kb+8..kb+15, kb+24..kb+31}; M = lane & 15.
__device__ __forceinline__ v16bf load_opA(const unsigned short* base, int stride,
                                          int lane, int kb) {
  const int r = lane & 15;
  const int h = (lane >> 4) & 1;
  const unsigned short* p = base + r * stride + kb + h * 8;
  OpU a;
  a.u[0] = *(const uint4*)(p);
  a.u[1] = *(const uint4*)(p + 16);
  return a.v;
}

// B-operand (Kx16, 16-bit elems). Column N = lane & 15 holds
// K {kb+8h..kb+8h+7, kb+16+8h..kb+16+8h+7}. Source tile is row-major [N][stride] over K.
__device__ __forceinline__ v16bf load_opB(const unsigned short* base, int stride,
                                          int lane, int kb) {
  const int n = lane & 15;
  const int h = (lane >> 4) & 1;
  const unsigned short* p = base + n * stride + kb + h * 8;
  OpU b;
  b.u[0] = *(const uint4*)(p);
  b.u[1] = *(const uint4*)(p + 16);
  return b.v;
}

struct __align__(16) SMem {
  unsigned short q[LBLK * DIM_E];      // (gain/tau)*Q rows for this l-block (bf16)
  unsigned short kneg[LBLK * DIM_E];   // -(gain/tau)*K rows for this l-block (bf16)
  unsigned short ks[SBLK * DIM_E];     // K rows for current s-block
  unsigned short qs[SBLK * DIM_E];     // Q rows for current s-block
  unsigned short vt[DIM_E * SBLK];     // V s-block, transposed [e][s]
  unsigned short att[8 * 16 * SBLK];   // per-wave att staging (16 x 32)
};

__global__ __launch_bounds__(256) void lie_attn_fused(
    const float* __restrict__ Q, const float* __restrict__ K,
    const float* __restrict__ V, const float* __restrict__ plog_gain,
    const float* __restrict__ plog_tau, float* __restrict__ Vout,
    float* __restrict__ Aout, float* __restrict__ Ent) {
  __shared__ SMem sm;
  const int tid  = threadIdx.x;
  const int lane = tid & 31;
  const int wave = tid >> 5;
  const int bh   = blockIdx.y;            // b*16 + h
  const int b    = bh >> 4;
  const int h    = bh & 15;
  const int lbase = blockIdx.x * LBLK;

  const float gain = fminf(fmaxf(expf(plog_gain[0]), 1e-3f), 1000.0f);
  const float tau  = fminf(fmaxf(expf(plog_tau[0]),  1e-3f), 10.0f);
  const float gt   = gain / tau;

  const size_t pitch = (size_t)DIM_H * DIM_E;   // 1024 floats between rows
  const float* Qb = Q + (size_t)b * DIM_L * pitch + (size_t)h * DIM_E;
  const float* Kb = K + (size_t)b * DIM_S * pitch + (size_t)h * DIM_E;
  const float* Vb = V + (size_t)b * DIM_S * pitch + (size_t)h * DIM_E;

  // Stage (gt*Q_l) and (-gt*K_l) for the whole l-block as bf16.
  // Folding gt here applies it once per l-row instead of once per score element.
  for (int idx = tid; idx < LBLK * (DIM_E / 4); idx += 256) {
    const int row = idx >> 4;
    const int c   = (idx & 15) * 4;
    const size_t go = (size_t)(lbase + row) * pitch + c;
    const float4 qv = *(const float4*)(Qb + go);
    sm.q[row * 64 + c + 0] = f2bf(qv.x * gt);
    sm.q[row * 64 + c + 1] = f2bf(qv.y * gt);
    sm.q[row * 64 + c + 2] = f2bf(qv.z * gt);
    sm.q[row * 64 + c + 3] = f2bf(qv.w * gt);
    const float4 kv = *(const float4*)(Kb + go);
    sm.kneg[row * 64 + c + 0] = f2bf(-kv.x * gt);
    sm.kneg[row * 64 + c + 1] = f2bf(-kv.y * gt);
    sm.kneg[row * 64 + c + 2] = f2bf(-kv.z * gt);
    sm.kneg[row * 64 + c + 3] = f2bf(-kv.w * gt);
  }
  __syncthreads();

  // Per-wave invariant A-operands (E=64 -> two K=32 chunks each).
  const unsigned short* qtile = sm.q    + wave * 16 * 64;
  const unsigned short* ktile = sm.kneg + wave * 16 * 64;
  const v16bf aq0 = load_opA(qtile, 64, lane, 0);
  const v16bf aq1 = load_opA(qtile, 64, lane, 32);
  const v16bf ak0 = load_opA(ktile, 64, lane, 0);
  const v16bf ak1 = load_opA(ktile, 64, lane, 32);

  v8f vacc[4];
#pragma unroll
  for (int j = 0; j < 4; ++j) {
    v8f z = {0.f, 0.f, 0.f, 0.f, 0.f, 0.f, 0.f, 0.f};
    vacc[j] = z;
  }
  float ent[8] = {0.f, 0.f, 0.f, 0.f, 0.f, 0.f, 0.f, 0.f};

  const int n  = lane & 15;        // C/D column within tile
  const int hh = lane >> 4;        // C/D row-half
  unsigned short* attw = sm.att + wave * (16 * SBLK);

  for (int it = 0; it < DIM_S / SBLK; ++it) {
    const int sbase = it * SBLK;

    // Cooperative staging of K_s, Q_s (row-major) and V_s (transposed).
    for (int idx = tid; idx < SBLK * (DIM_E / 4); idx += 256) {
      const int row = idx >> 4;
      const int c   = (idx & 15) * 4;
      const size_t go = (size_t)(sbase + row) * pitch + c;
      const float4 kv = *(const float4*)(Kb + go);
      sm.ks[row * 64 + c + 0] = f2bf(kv.x);
      sm.ks[row * 64 + c + 1] = f2bf(kv.y);
      sm.ks[row * 64 + c + 2] = f2bf(kv.z);
      sm.ks[row * 64 + c + 3] = f2bf(kv.w);
      const float4 qv = *(const float4*)(Qb + go);
      sm.qs[row * 64 + c + 0] = f2bf(qv.x);
      sm.qs[row * 64 + c + 1] = f2bf(qv.y);
      sm.qs[row * 64 + c + 2] = f2bf(qv.z);
      sm.qs[row * 64 + c + 3] = f2bf(qv.w);
      const float4 vv = *(const float4*)(Vb + go);
      sm.vt[(c + 0) * SBLK + row] = f2bf(vv.x);
      sm.vt[(c + 1) * SBLK + row] = f2bf(vv.y);
      sm.vt[(c + 2) * SBLK + row] = f2bf(vv.z);
      sm.vt[(c + 3) * SBLK + row] = f2bf(vv.w);
      if (it + 1 < DIM_S / SBLK) {
        const size_t pg = go + (size_t)SBLK * pitch;
        __builtin_prefetch(Kb + pg, 0, 0);   // global_prefetch_b8
        __builtin_prefetch(Qb + pg, 0, 0);
        __builtin_prefetch(Vb + pg, 0, 0);
      }
    }
    __syncthreads();

    // Two 16x16 comm tiles: gt*comm = (gt*Q_l) K_s^T + (-gt*K_l) Q_s^T
#pragma unroll
    for (int sub = 0; sub < 2; ++sub) {
      const unsigned short* ksub = sm.ks + sub * 16 * 64;
      const unsigned short* qsub = sm.qs + sub * 16 * 64;
      const v16bf bk0 = load_opB(ksub, 64, lane, 0);
      const v16bf bk1 = load_opB(ksub, 64, lane, 32);
      const v16bf bq0 = load_opB(qsub, 64, lane, 0);
      const v16bf bq1 = load_opB(qsub, 64, lane, 32);
      v8f acc = {0.f, 0.f, 0.f, 0.f, 0.f, 0.f, 0.f, 0.f};
      acc = __builtin_amdgcn_wmma_f32_16x16x32_bf16(false, aq0, false, bk0,
                                                    (short)0, acc, false, false);
      acc = __builtin_amdgcn_wmma_f32_16x16x32_bf16(false, aq1, false, bk1,
                                                    (short)0, acc, false, false);
      acc = __builtin_amdgcn_wmma_f32_16x16x32_bf16(false, ak0, false, bq0,
                                                    (short)0, acc, false, false);
      acc = __builtin_amdgcn_wmma_f32_16x16x32_bf16(false, ak1, false, bq1,
                                                    (short)0, acc, false, false);

      // Pointwise: att = (1/8)*gelu_exact(tanh(gt*comm)); write A, entropy, stage bf16.
      const int scol = sbase + sub * 16 + n;
      const size_t abase =
          ((size_t)bh * DIM_L + (size_t)(lbase + wave * 16 + hh * 8)) * DIM_S + scol;
#pragma unroll
      for (int v = 0; v < 8; ++v) {
        const float t   = fast_tanh(acc[v]);
        const float att = att_from_t(t);
        Aout[abase + (size_t)v * DIM_S] = att;
        ent[v] += att * __logf(fmaxf(att, 1e-8f));
        attw[(v + hh * 8) * SBLK + sub * 16 + n] = f2bf(att);
      }
    }

    // Second GEMM: Vout_tile += att(16x32) x V_s(32x64), 4 N-tiles of 16.
    const v16bf aatt = load_opA(attw, SBLK, lane, 0);
#pragma unroll
    for (int j = 0; j < 4; ++j) {
      const v16bf bv = load_opB(sm.vt + j * 16 * SBLK, SBLK, lane, 0);
      vacc[j] = __builtin_amdgcn_wmma_f32_16x16x32_bf16(false, aatt, false, bv,
                                                        (short)0, vacc[j], false, false);
    }
    __syncthreads();   // protect s-block LDS before next iteration overwrites
  }

  // Write V output: (b, l, h, e)
  const int l0 = lbase + wave * 16 + hh * 8;
#pragma unroll
  for (int j = 0; j < 4; ++j) {
#pragma unroll
    for (int v = 0; v < 8; ++v) {
      const size_t o =
          ((size_t)b * DIM_L + (size_t)(l0 + v)) * pitch + (size_t)h * DIM_E + j * 16 + n;
      Vout[o] = vacc[j][v];
    }
  }

  // Entropy: reduce across the 16 lanes sharing each row (wave32 halves).
#pragma unroll
  for (int v = 0; v < 8; ++v) {
    float e = ent[v];
    e += __shfl_xor(e, 1, 32);
    e += __shfl_xor(e, 2, 32);
    e += __shfl_xor(e, 4, 32);
    e += __shfl_xor(e, 8, 32);
    if (n == 0) Ent[(size_t)bh * DIM_L + l0 + v] = -e;
  }
}

extern "C" void kernel_launch(void* const* d_in, const int* in_sizes, int n_in,
                              void* d_out, int out_size, void* d_ws, size_t ws_size,
                              hipStream_t stream) {
  (void)in_sizes; (void)n_in; (void)d_ws; (void)ws_size; (void)out_size;
  const float* q  = (const float*)d_in[0];
  const float* k  = (const float*)d_in[1];
  const float* v  = (const float*)d_in[2];
  const float* lg = (const float*)d_in[3];
  const float* lt = (const float*)d_in[4];

  float* out  = (float*)d_out;
  float* Vout = out;                                               // B*L*H*E
  float* Aout = Vout + (size_t)DIM_B * DIM_L * DIM_H * DIM_E;      // B*H*L*S
  float* Ent  = Aout + (size_t)DIM_B * DIM_H * DIM_L * DIM_S;      // B*H*L

  dim3 grid(DIM_L / LBLK, DIM_B * DIM_H);
  dim3 block(256);
  lie_attn_fused<<<grid, block, 0, stream>>>(q, k, v, lg, lt, Vout, Aout, Ent);
}
